// RNN_73813307949430
// MI455X (gfx1250) — matrix operand order repaired and checked by reading, not compiled, not run on hardware
//
#include <hip/hip_runtime.h>

typedef __attribute__((ext_vector_type(16))) _Float16 v16h;
typedef __attribute__((ext_vector_type(8)))  _Float16 v8h;
typedef __attribute__((ext_vector_type(8)))  float    v8f;

#define HID 50
#define NSTEP 10
#define FIN 5
#define NG 200           // 4*HID
#define NPAD 208         // 13 tiles of 16
#define KPAD 64
#define NTILES 13
#define STRD 56          // seq halfs per timestep
#define WAVES 8
#define THREADS 256
#define RPW 16           // rows per wave
#define RPB 128          // rows per block

// ---- LDS layout (bytes) ----
#define OFF_WT   0
#define SZ_WMAT  (NPAD*KPAD*2)                 // 26624
#define OFF_UT   (OFF_WT + SZ_WMAT)            // 26624
#define OFF_SEQ  (OFF_UT + SZ_WMAT)            // 53248
#define SZ_SEQW  (RPW*NSTEP*STRD*2)            // 17920 per wave
#define OFF_ZB   (OFF_SEQ + WAVES*SZ_SEQW)     // 196608
#define SZ_ZBW   (RPW*NPAD*4)                  // 13312 per wave
#define OFF_LB   (OFF_ZB + WAVES*SZ_ZBW)       // 303104
#define SZ_LB    (4*NPAD*4)                    // 3328 (bias padded to 208 cols)
#define OFF_WD1  (OFF_LB + SZ_LB)              // 306432
#define OFF_BD1  (OFF_WD1 + HID*HID*4)         // 316432
#define OFF_WD2  (OFF_BD1 + HID*4)             // 316632
#define OFF_BD2  (OFF_WD2 + HID*4)             // 316832
#define OFF_ZERO (OFF_BD2 + 16)                // 316848 (16B aligned)
#define LDS_BYTES (OFF_ZERO + 16)              // 316864 < 320KB

#define WMMA_F16(A, B, C) \
    __builtin_amdgcn_wmma_f32_16x16x32_f16(false, (A), false, (B), (short)0, (C), false, false)

// gfx1250 has a native TANH trans op; 1 trans instead of exp+rcp (2 trans).
__device__ __forceinline__ float fast_tanh(float x) {
#if __has_builtin(__builtin_amdgcn_tanhf)
    return __builtin_amdgcn_tanhf(x);
#else
    return 2.0f / (1.0f + __expf(-2.0f * x)) - 1.0f;
#endif
}
__device__ __forceinline__ float fast_sigmoid(float x) {
    return 0.5f + 0.5f * fast_tanh(0.5f * x);
}

// B fragment (32x16 f16): lane holds column n = nt*16+(lane&15);
// halfs j -> k = ks*32 + (lane>>4)*16 + j  => one 32B contiguous aligned read.
__device__ __forceinline__ v16h load_b_frag(const _Float16* wt, int nt, int ks, int lane) {
    const int nl = lane & 15, hb = lane >> 4;
    return *(const v16h*)(wt + (size_t)(nt * 16 + nl) * KPAD + ks * 32 + hb * 16);
}

// A fragment (16x32 f16) from one seq row. Branchless: out-of-row pad group
// reads a dedicated zeroed LDS region via address select (no EXEC games).
__device__ __forceinline__ v16h load_a_frag(const _Float16* rowt, const _Float16* zpad,
                                            int ks, int hb) {
    const int o0 = ks * 32 + 16 * hb;
    const int o1 = o0 + 8;
    const v8h lo = *(const v8h*)(rowt + o0);
    const _Float16* p1 = (o1 < STRD) ? (rowt + o1) : zpad;
    const v8h hi = *(const v8h*)p1;
    return __builtin_shufflevector(lo, hi, 0,1,2,3,4,5,6,7,8,9,10,11,12,13,14,15);
}

// One timestep's z = bias + x@W (+ h@U); 13 N-tiles in pairs with two
// independent accumulator chains; bias pre-splatted into the C operand.
template <bool XK2, bool HASH>
__device__ __forceinline__ void gemm_step(const _Float16* __restrict__ WT,
                                          const _Float16* __restrict__ UT,
                                          float* __restrict__ zb,
                                          const float* __restrict__ bv,
                                          v16h ax0, v16h ax1, v16h ah0, v16h ah1,
                                          int lane, int ml, int hb) {
    const int zbase = (8 * hb) * NPAD + ml;
#pragma unroll
    for (int p = 0; p < 6; ++p) {
        const int n0 = 2 * p, n1 = 2 * p + 1;
        v8f a0, a1;
#pragma unroll
        for (int r = 0; r < 8; ++r) { a0[r] = bv[n0]; a1[r] = bv[n1]; }
        {
            v16h b0 = load_b_frag(WT, n0, 0, lane);
            v16h b1 = load_b_frag(WT, n1, 0, lane);
            a0 = WMMA_F16(ax0, b0, a0);
            a1 = WMMA_F16(ax0, b1, a1);
        }
        if constexpr (XK2) {
            v16h b0 = load_b_frag(WT, n0, 1, lane);
            v16h b1 = load_b_frag(WT, n1, 1, lane);
            a0 = WMMA_F16(ax1, b0, a0);
            a1 = WMMA_F16(ax1, b1, a1);
        }
        if constexpr (HASH) {
            v16h b0 = load_b_frag(UT, n0, 0, lane);
            v16h b1 = load_b_frag(UT, n1, 0, lane);
            a0 = WMMA_F16(ah0, b0, a0);
            a1 = WMMA_F16(ah0, b1, a1);
            v16h c0 = load_b_frag(UT, n0, 1, lane);
            v16h c1 = load_b_frag(UT, n1, 1, lane);
            a0 = WMMA_F16(ah1, c0, a0);
            a1 = WMMA_F16(ah1, c1, a1);
        }
#pragma unroll
        for (int r = 0; r < 8; ++r) {
            zb[zbase + r * NPAD + n0 * 16] = a0[r];
            zb[zbase + r * NPAD + n1 * 16] = a1[r];
        }
    }
    {   // tail tile 12
        const int n0 = 12;
        v8f a0;
#pragma unroll
        for (int r = 0; r < 8; ++r) a0[r] = bv[n0];
        a0 = WMMA_F16(ax0, load_b_frag(WT, n0, 0, lane), a0);
        if constexpr (XK2) a0 = WMMA_F16(ax1, load_b_frag(WT, n0, 1, lane), a0);
        if constexpr (HASH) {
            a0 = WMMA_F16(ah0, load_b_frag(UT, n0, 0, lane), a0);
            a0 = WMMA_F16(ah1, load_b_frag(UT, n0, 1, lane), a0);
        }
#pragma unroll
        for (int r = 0; r < 8; ++r) zb[zbase + r * NPAD + n0 * 16] = a0[r];
    }
}

// i/f/g/o nonlinearities + c/h update; bias already in z; addresses precomputed.
__device__ __forceinline__ void gate_update(const float* __restrict__ zb,
                                            _Float16* __restrict__ seqw,
                                            const int* __restrict__ zoff,
                                            const int* __restrict__ soff,
                                            float c[25], int tOff) {
#pragma unroll
    for (int q = 0; q < 25; ++q) {
        const float zi = zb[zoff[q]];
        const float zf = zb[zoff[q] + 50];
        const float zg = zb[zoff[q] + 100];
        const float zo = zb[zoff[q] + 150];
        const float ig = fast_sigmoid(zi);
        const float fg = fast_sigmoid(zf);
        const float gg = fast_tanh(zg);
        const float og = fast_sigmoid(zo);
        const float cn = fg * c[q] + ig * gg;
        c[q] = cn;
        seqw[soff[q] + tOff] = (_Float16)(og * fast_tanh(cn));
    }
}

__global__ __launch_bounds__(THREADS, 2)
void lstm4_wmma_kernel(const float* __restrict__ x,
                       const float* __restrict__ W1, const float* __restrict__ U1, const float* __restrict__ b1,
                       const float* __restrict__ W2, const float* __restrict__ U2, const float* __restrict__ b2,
                       const float* __restrict__ W3, const float* __restrict__ U3, const float* __restrict__ b3,
                       const float* __restrict__ W4, const float* __restrict__ U4, const float* __restrict__ b4,
                       const float* __restrict__ Wd1, const float* __restrict__ bd1,
                       const float* __restrict__ Wd2, const float* __restrict__ bd2,
                       float* __restrict__ out, int bTot) {
    extern __shared__ char smem[];
    _Float16* WT   = (_Float16*)(smem + OFF_WT);
    _Float16* UT   = (_Float16*)(smem + OFF_UT);
    float*    LB   = (float*)(smem + OFF_LB);
    float*    WD1s = (float*)(smem + OFF_WD1);
    float*    BD1s = (float*)(smem + OFF_BD1);
    float*    WD2s = (float*)(smem + OFF_WD2);
    float*    BD2s = (float*)(smem + OFF_BD2);
    _Float16* ZPAD = (_Float16*)(smem + OFF_ZERO);

    const int tid  = threadIdx.x;
    const int wv   = tid >> 5;
    const int lane = tid & 31;
    const int ml   = lane & 15;
    const int hb   = lane >> 4;

    _Float16* seqw = (_Float16*)(smem + OFF_SEQ + (size_t)wv * SZ_SEQW);
    float*    zb   = (float*)(smem + OFF_ZB + (size_t)wv * SZ_ZBW);
    const int row0 = blockIdx.x * RPB + wv * RPW;

    // ---- one-time cooperative loads / zero-init ----
    {
        const float* bs[4] = {b1, b2, b3, b4};
        for (int i = tid; i < 4 * NPAD; i += THREADS) {
            const int li = i / NPAD, n = i % NPAD;
            LB[i] = (n < NG) ? bs[li][n] : 0.0f;
        }
        for (int i = tid; i < HID * HID; i += THREADS) WD1s[i] = Wd1[i];
        for (int i = tid; i < HID; i += THREADS) { BD1s[i] = bd1[i]; WD2s[i] = Wd2[i]; }
        if (tid == 0) BD2s[0] = bd2[0];
        for (int i = tid; i < 8; i += THREADS) ZPAD[i] = (_Float16)0.0f;
        _Float16* s0 = (_Float16*)(smem + OFF_SEQ);
        for (int i = tid; i < WAVES * SZ_SEQW / 2; i += THREADS) s0[i] = (_Float16)0.0f;
    }

    // per-lane constant addresses for the gate phase (element e = lane + 32q)
    int zoff[25], soff[25];
#pragma unroll
    for (int q = 0; q < 25; ++q) {
        const int e = lane + 32 * q;
        const int m = e / HID, j = e % HID;
        zoff[q] = m * NPAD + j;
        soff[q] = m * NSTEP * STRD + j;
    }

    const float* Ws[4] = {W1, W2, W3, W4};
    const float* Us[4] = {U1, U2, U3, U4};
    float c[25];
    const v16h vzero = {};

    auto build_x_frag = [&](int t) -> v16h {
        v16h a = {};
        if (hb == 0 && (row0 + ml) < bTot) {
            const float* xp = x + ((size_t)(row0 + ml) * NSTEP + t) * FIN;
#pragma unroll
            for (int k = 0; k < FIN; ++k) a[k] = (_Float16)xp[k];
        }
        return a;
    };

    for (int l = 0; l < 4; ++l) {
        __syncthreads();  // everyone done with previous layer's weights
        const int kin = (l == 0) ? FIN : HID;
        const float* Wg = Ws[l];
        const float* Ug = Us[l];
        for (int i = tid; i < NPAD * KPAD; i += THREADS) {
            const int n = i / KPAD, k = i % KPAD;
            WT[i] = (n < NG && k < kin) ? (_Float16)Wg[k * NG + n] : (_Float16)0.0f;
            UT[i] = (n < NG && k < HID) ? (_Float16)Ug[k * NG + n] : (_Float16)0.0f;
        }
        __syncthreads();

        // per-tile bias value for this layer (column = nt*16 + ml, lane-uniform in D layout)
        float bv[NTILES];
#pragma unroll
        for (int nt = 0; nt < NTILES; ++nt) bv[nt] = LB[l * NPAD + nt * 16 + ml];

#pragma unroll
        for (int q = 0; q < 25; ++q) c[q] = 0.0f;

        if (l == 0) {
            gemm_step<false, false>(WT, UT, zb, bv, build_x_frag(0), vzero, vzero, vzero,
                                    lane, ml, hb);
            gate_update(zb, seqw, zoff, soff, c, 0);
            for (int t = 1; t < NSTEP; ++t) {
                const _Float16* rowp = seqw + (size_t)(ml * NSTEP + (t - 1)) * STRD;
                v16h ah0 = load_a_frag(rowp, ZPAD, 0, hb);
                v16h ah1 = load_a_frag(rowp, ZPAD, 1, hb);
                gemm_step<false, true>(WT, UT, zb, bv, build_x_frag(t), vzero, ah0, ah1,
                                       lane, ml, hb);
                gate_update(zb, seqw, zoff, soff, c, t * STRD);
            }
        } else {
            {   // t = 0
                const _Float16* rowt = seqw + (size_t)(ml * NSTEP + 0) * STRD;
                v16h ax0 = load_a_frag(rowt, ZPAD, 0, hb);
                v16h ax1 = load_a_frag(rowt, ZPAD, 1, hb);
                gemm_step<true, false>(WT, UT, zb, bv, ax0, ax1, vzero, vzero, lane, ml, hb);
                gate_update(zb, seqw, zoff, soff, c, 0);
            }
            for (int t = 1; t < NSTEP; ++t) {
                const _Float16* rowt = seqw + (size_t)(ml * NSTEP + t) * STRD;
                v16h ax0 = load_a_frag(rowt, ZPAD, 0, hb);
                v16h ax1 = load_a_frag(rowt, ZPAD, 1, hb);
                const _Float16* rowp = seqw + (size_t)(ml * NSTEP + (t - 1)) * STRD;
                v16h ah0 = load_a_frag(rowp, ZPAD, 0, hb);
                v16h ah1 = load_a_frag(rowp, ZPAD, 1, hb);
                gemm_step<true, true>(WT, UT, zb, bv, ax0, ax1, ah0, ah1, lane, ml, hb);
                gate_update(zb, seqw, zoff, soff, c, t * STRD);
            }
        }
    }

    // ---- dense head: out = sigmoid((h@Wd1 + bd1)@Wd2 + bd2), h = seq[t=9] ----
    if (lane < 16 && (row0 + ml) < bTot) {
        const _Float16* hp = seqw + (size_t)(ml * NSTEP + (NSTEP - 1)) * STRD;
        float h[HID];
#pragma unroll
        for (int k = 0; k < HID; ++k) h[k] = (float)hp[k];
        float s2 = BD2s[0];
        for (int j = 0; j < HID; ++j) {
            float a = BD1s[j];
#pragma unroll 10
            for (int k = 0; k < HID; ++k) a += h[k] * WD1s[k * HID + j];
            s2 += a * WD2s[j];
        }
        out[row0 + ml] = fast_sigmoid(s2);
    }
}

extern "C" void kernel_launch(void* const* d_in, const int* in_sizes, int n_in,
                              void* d_out, int out_size, void* d_ws, size_t ws_size,
                              hipStream_t stream) {
    (void)n_in; (void)d_ws; (void)ws_size; (void)in_sizes;
    const float* x   = (const float*)d_in[0];
    const float* W1  = (const float*)d_in[1];
    const float* U1  = (const float*)d_in[2];
    const float* b1  = (const float*)d_in[3];
    const float* W2  = (const float*)d_in[4];
    const float* U2  = (const float*)d_in[5];
    const float* b2  = (const float*)d_in[6];
    const float* W3  = (const float*)d_in[7];
    const float* U3  = (const float*)d_in[8];
    const float* b3  = (const float*)d_in[9];
    const float* W4  = (const float*)d_in[10];
    const float* U4  = (const float*)d_in[11];
    const float* b4  = (const float*)d_in[12];
    const float* Wd1 = (const float*)d_in[13];
    const float* bd1 = (const float*)d_in[14];
    const float* Wd2 = (const float*)d_in[15];
    const float* bd2 = (const float*)d_in[16];
    float* out = (float*)d_out;

    const int bTot = out_size;                 // [B,1] f32
    const int blocks = (bTot + RPB - 1) / RPB; // 1024 for B=131072

    (void)hipFuncSetAttribute((const void*)lstm4_wmma_kernel,
                              hipFuncAttributeMaxDynamicSharedMemorySize, LDS_BYTES);

    lstm4_wmma_kernel<<<blocks, THREADS, LDS_BYTES, stream>>>(
        x, W1, U1, b1, W2, U2, b2, W3, U3, b3, W4, U4, b4,
        Wd1, bd1, Wd2, bd2, out, bTot);
}
